// Speller_56925496541926
// MI455X (gfx1250) — matrix operand order repaired and checked by reading, not compiled, not run on hardware
//
#include <hip/hip_runtime.h>

// ---------------------------------------------------------------------------
// Speller (LAS decoder), MI455X/gfx1250.
// 256 sequential steps, 4 kernels/step:
//   lstm_cell(1) -> attn(+y-pack for t+1) -> lstm_cell(2) -> dense_softmax
// GEMMs: bf16 WMMA (v_wmma_f32_16x16x32_bf16), fp32 accumulate, gates fused
// into the GEMM epilogue (one wave owns all 4 gate tiles of a u-slice).
// Attention: one-pass online softmax, streams encoder memory (bf16 if the
// workspace allows, else fp32) exactly once per step; it is L2-resident
// (64-128MB < 192MB L2) so HBM sees it ~once total.
// ---------------------------------------------------------------------------

#define UNITS 512
#define VOCAB 46
#define BATCH 64
#define TDEC  256
#define TENC  1024
#define K1    1088   // 1070 padded to multiple of 32
#define K2    1024
#define NOUT  2048   // 4*UNITS

typedef __bf16 bf16_t;
typedef __attribute__((ext_vector_type(16))) bf16_t v16bf;
typedef __attribute__((ext_vector_type(8)))  float  v8f;

__device__ __forceinline__ unsigned short f32_to_bf16(float f) {
    unsigned u = __float_as_uint(f);
    u += 0x7FFFu + ((u >> 16) & 1u);      // round-to-nearest-even
    return (unsigned short)(u >> 16);
}

// A fragment (16x32 bf16, row-major src; ISA 16-bit A layout):
// lane<16 : row=lane,    K = k0+[0..7]  and k0+16+[0..7]
// lane>=16: row=lane-16, K = k0+8+[0..7] and k0+24+[0..7]
__device__ __forceinline__ v16bf load_frag_a(const unsigned short* A, int lda,
                                             int row0, int k0, int lane) {
    int half = lane >> 4, r = lane & 15;
    const unsigned short* p = A + (size_t)(row0 + r) * lda + k0 + half * 8;
    union { v16bf v; uint4 q[2]; } f;
    f.q[0] = *(const uint4*)(p);
    f.q[1] = *(const uint4*)(p + 16);
    return f.v;
}

// B fragment (32x16 bf16) from pre-transposed weights Wt[N][K]:
// lane<16 : col=lane,    K = k0+[0..7] and k0+16+[0..7] ;  lane>=16: col=lane-16, K = k0+8+[0..7] and k0+24+[0..7]
__device__ __forceinline__ v16bf load_frag_b(const unsigned short* Wt, int ldb,
                                             int col0, int k0, int lane) {
    int half = lane >> 4, n = lane & 15;
    const unsigned short* p = Wt + (size_t)(col0 + n) * ldb + k0 + half * 8;
    union { v16bf v; uint4 q[2]; } f;
    f.q[0] = *(const uint4*)(p);
    f.q[1] = *(const uint4*)(p + 16);
    return f.v;
}

#define WMMA_BF16(a, b, c) \
    __builtin_amdgcn_wmma_f32_16x16x32_bf16(false, (a), false, (b), (short)0, (c), false, false)

// Fused LSTM cell: z = X(bf16)[64,K] @ Wt(bf16)[2048,K]^T, gates, state update.
// 128 tiles (4 batch-tiles x 32 u-tiles); wave w owns one (tm,tu) and computes
// the 4 gate tiles i/f/g/o sharing one A fragment per K-step (4 WMMA / k0).
// Reads Xin (this step's packed activations), writes bf16(h) into Xnext slot.
__global__ __launch_bounds__(256)
void lstm_cell(const unsigned short* __restrict__ Xin, int K,
               const unsigned short* __restrict__ Wt,
               const float* __restrict__ bias,
               float* __restrict__ C, float* __restrict__ H,
               unsigned short* __restrict__ Xnext, int xld, int xoff) {
    int wave = threadIdx.x >> 5;
    int lane = threadIdx.x & 31;
    int tile = blockIdx.x * 8 + wave;       // 16 blocks * 8 waves = 128 tiles
    int tm = tile >> 5, tu = tile & 31;

    v8f ai = {}, af = {}, ag = {}, ao = {};
    for (int k0 = 0; k0 < K; k0 += 32) {
        v16bf a  = load_frag_a(Xin, K, tm * 16, k0, lane);
        v16bf bi = load_frag_b(Wt, K, tu * 16,             k0, lane);
        v16bf bf = load_frag_b(Wt, K, UNITS   + tu * 16,   k0, lane);
        v16bf bg = load_frag_b(Wt, K, 2*UNITS + tu * 16,   k0, lane);
        v16bf bo = load_frag_b(Wt, K, 3*UNITS + tu * 16,   k0, lane);
        ai = WMMA_BF16(a, bi, ai);
        af = WMMA_BF16(a, bf, af);
        ag = WMMA_BF16(a, bg, ag);
        ao = WMMA_BF16(a, bo, ao);
    }
    // C/D layout: lane<16 -> N=lane, M=v ; lane>=16 -> N=lane-16, M=8+v
    int half = lane >> 4, n = lane & 15;
    int u = tu * 16 + n;
    float b_i = bias[u], b_f = bias[UNITS + u], b_g = bias[2*UNITS + u], b_o = bias[3*UNITS + u];
#pragma unroll
    for (int v = 0; v < 8; ++v) {
        int b   = tm * 16 + half * 8 + v;
        size_t idx = (size_t)b * UNITS + u;
        float ig = 1.f / (1.f + __expf(-(ai[v] + b_i)));
        float fg = 1.f / (1.f + __expf(-(af[v] + b_f)));
        float gg = tanhf(ag[v] + b_g);
        float og = 1.f / (1.f + __expf(-(ao[v] + b_o)));
        float cn = fg * C[idx] + ig * gg;
        float hn = og * tanhf(cn);
        C[idx] = cn;
        H[idx] = hn;
        Xnext[(size_t)b * xld + xoff + u] = f32_to_bf16(hn);
    }
}

// One-pass online-softmax dot attention; one block (8 waves) per batch elem.
// Also packs ctx (bf16) into next-step X1 / current-step X2, and y(t+1) into
// next-step X1 (replaces a separate pack kernel).
template <bool BF16H>
__global__ __launch_bounds__(256)
void attn_kernel(const void* __restrict__ Hptr, const float* __restrict__ S1,
                 float* __restrict__ Ctx,
                 unsigned short* __restrict__ X1n, unsigned short* __restrict__ X2c,
                 const float* __restrict__ Y, int t) {
    int b    = blockIdx.x;
    int wave = threadIdx.x >> 5, lane = threadIdx.x & 31;
    const float* s = S1 + (size_t)b * UNITS;

    float sreg[16], acc[16];
#pragma unroll
    for (int j = 0; j < 16; ++j) { sreg[j] = s[lane * 16 + j]; acc[j] = 0.f; }
    float m = -1e30f, l = 0.f;

    for (int te = wave; te < TENC; te += 8) {
        float hv[16];
        if (BF16H) {
            const unsigned short* hr = (const unsigned short*)Hptr
                + ((size_t)b * TENC + te) * UNITS + lane * 16;
            uint4 q0 = *(const uint4*)hr;
            uint4 q1 = *(const uint4*)(hr + 8);
            unsigned q[8] = {q0.x, q0.y, q0.z, q0.w, q1.x, q1.y, q1.z, q1.w};
#pragma unroll
            for (int j = 0; j < 8; ++j) {
                hv[2*j]   = __uint_as_float(q[j] << 16);
                hv[2*j+1] = __uint_as_float(q[j] & 0xFFFF0000u);
            }
        } else {
            const float* hr = (const float*)Hptr + ((size_t)b * TENC + te) * UNITS + lane * 16;
#pragma unroll
            for (int j = 0; j < 16; ++j) hv[j] = hr[j];
        }
        float p = 0.f;
#pragma unroll
        for (int j = 0; j < 16; ++j) p = fmaf(sreg[j], hv[j], p);
#pragma unroll
        for (int off = 16; off >= 1; off >>= 1) p += __shfl_xor(p, off, 32);
        float mn = fmaxf(m, p);
        float al = __expf(m - mn);
        float w  = __expf(p - mn);
        l = l * al + w;
#pragma unroll
        for (int j = 0; j < 16; ++j) acc[j] = acc[j] * al + w * hv[j];
        m = mn;
    }

    __shared__ float sm[8], sl[8], sacc[8][UNITS];
    if (lane == 0) { sm[wave] = m; sl[wave] = l; }
#pragma unroll
    for (int j = 0; j < 16; ++j) sacc[wave][lane * 16 + j] = acc[j];
    __syncthreads();

    float M = -1e30f;
#pragma unroll
    for (int w = 0; w < 8; ++w) M = fmaxf(M, sm[w]);
    float L = 0.f;
#pragma unroll
    for (int w = 0; w < 8; ++w) L += sl[w] * __expf(sm[w] - M);
    float invL = 1.f / L;

    for (int u = threadIdx.x; u < UNITS; u += 256) {
        float a = 0.f;
#pragma unroll
        for (int w = 0; w < 8; ++w) a += sacc[w][u] * __expf(sm[w] - M);
        float cv = a * invL;
        Ctx[(size_t)b * UNITS + u] = cv;
        unsigned short bv = f32_to_bf16(cv);
        X1n[(size_t)b * K1 + UNITS + VOCAB + u] = bv;   // ctx slice of next x1
        X2c[(size_t)b * K2 + u]                 = bv;   // ctx slice of this x2
    }
    if (t + 1 < TDEC && threadIdx.x < VOCAB) {
        int v = threadIdx.x;
        X1n[(size_t)b * K1 + UNITS + v] =
            f32_to_bf16(Y[((size_t)b * TDEC + (t + 1)) * VOCAB + v]);
    }
}

// logits = [h2, ctx] @ Wd + bd, softmax over VOCAB=46.
__global__ void dense_softmax(const float* __restrict__ H2, const float* __restrict__ Ctx,
                              const float* __restrict__ Wd, const float* __restrict__ bd,
                              float* __restrict__ Out) {
    int b = blockIdx.x;
    int v = threadIdx.x;                  // blockDim = 64
    __shared__ float lg[VOCAB];
    if (v < VOCAB) {
        float acc = bd[v];
        const float* h2 = H2  + (size_t)b * UNITS;
        const float* cx = Ctx + (size_t)b * UNITS;
        for (int k = 0; k < UNITS; ++k) acc = fmaf(h2[k], Wd[(size_t)k * VOCAB + v], acc);
        for (int k = 0; k < UNITS; ++k) acc = fmaf(cx[k], Wd[(size_t)(UNITS + k) * VOCAB + v], acc);
        lg[v] = acc;
    }
    __syncthreads();
    if (v < VOCAB) {
        float M = -1e30f;
        for (int j = 0; j < VOCAB; ++j) M = fmaxf(M, lg[j]);
        float S = 0.f;
        for (int j = 0; j < VOCAB; ++j) S += __expf(lg[j] - M);
        Out[(size_t)b * VOCAB + v] = __expf(lg[v] - M) / S;
    }
}

// --- prologue kernels (run every call: deterministic, no cached state) -----

// W1t[n][k] = bf16( k<512 ? W1[k][n]+U1[k][n] : k<1070 ? W1[k][n] : 0 )
__global__ void prep_w1(const float* __restrict__ W1, const float* __restrict__ U1,
                        unsigned short* __restrict__ W1t) {
    int idx = blockIdx.x * 256 + threadIdx.x;
    if (idx >= NOUT * K1) return;
    int n = idx / K1, k = idx % K1;
    float v;
    if (k < UNITS)                      v = W1[(size_t)k * NOUT + n] + U1[(size_t)k * NOUT + n];
    else if (k < UNITS + VOCAB + UNITS) v = W1[(size_t)k * NOUT + n];
    else                                v = 0.f;
    W1t[idx] = f32_to_bf16(v);
}

// W2t[n][k] = bf16( k<512 ? W2[k][n] : U2[k-512][n] )   (x2 = [ctx, h2])
__global__ void prep_w2(const float* __restrict__ W2, const float* __restrict__ U2,
                        unsigned short* __restrict__ W2t) {
    int idx = blockIdx.x * 256 + threadIdx.x;
    if (idx >= NOUT * K2) return;
    int n = idx / K2, k = idx % K2;
    float v = (k < UNITS) ? W2[(size_t)k * NOUT + n]
                          : U2[(size_t)(k - UNITS) * NOUT + n];
    W2t[idx] = f32_to_bf16(v);
}

// Zero state + both activation buffers; pack y(0) into X1a's y slice.
__global__ void init_state(unsigned short* __restrict__ X1a, unsigned short* __restrict__ X1b,
                           unsigned short* __restrict__ X2a, unsigned short* __restrict__ X2b,
                           float* __restrict__ C1, float* __restrict__ C2,
                           const float* __restrict__ Y) {
    int idx = blockIdx.x * 256 + threadIdx.x;
    if (idx < BATCH * K1) {
        int b = idx / K1, k = idx % K1;
        unsigned short v0 = 0;
        if (k >= UNITS && k < UNITS + VOCAB)
            v0 = f32_to_bf16(Y[(size_t)b * TDEC * VOCAB + (k - UNITS)]);   // t = 0
        X1a[idx] = v0;
        X1b[idx] = 0;
    }
    if (idx < BATCH * K2) { X2a[idx] = 0; X2b[idx] = 0; }
    if (idx < BATCH * UNITS) { C1[idx] = 0.f; C2[idx] = 0.f; }
}

// fp32 encoder memory -> bf16 (halves attention's per-step L2 traffic).
__global__ void conv_h(const float* __restrict__ Hf, unsigned short* __restrict__ Hb) {
    size_t idx = ((size_t)blockIdx.x * 256 + threadIdx.x) * 4;
    float4 f = *(const float4*)(Hf + idx);
    ushort4 o;
    o.x = f32_to_bf16(f.x); o.y = f32_to_bf16(f.y);
    o.z = f32_to_bf16(f.z); o.w = f32_to_bf16(f.w);
    *(ushort4*)(Hb + idx) = o;
}

// ---------------------------------------------------------------------------

extern "C" void kernel_launch(void* const* d_in, const int* in_sizes, int n_in,
                              void* d_out, int out_size, void* d_ws, size_t ws_size,
                              hipStream_t stream) {
    (void)in_sizes; (void)n_in; (void)out_size;
    const float* h  = (const float*)d_in[0];
    const float* y  = (const float*)d_in[1];
    const float* W1 = (const float*)d_in[2];
    const float* U1 = (const float*)d_in[3];
    const float* b1 = (const float*)d_in[4];
    const float* W2 = (const float*)d_in[5];
    const float* U2 = (const float*)d_in[6];
    const float* b2 = (const float*)d_in[7];
    const float* Wd = (const float*)d_in[8];
    const float* bd = (const float*)d_in[9];
    float* out = (float*)d_out;

    size_t used = 0;
    auto alloc = [&](size_t bytes) {
        char* p = (char*)d_ws + used;
        used += (bytes + 255) & ~(size_t)255;
        return p;
    };
    unsigned short* W1t = (unsigned short*)alloc((size_t)NOUT * K1 * 2);
    unsigned short* W2t = (unsigned short*)alloc((size_t)NOUT * K2 * 2);
    unsigned short* X1[2] = { (unsigned short*)alloc((size_t)BATCH * K1 * 2),
                              (unsigned short*)alloc((size_t)BATCH * K1 * 2) };
    unsigned short* X2[2] = { (unsigned short*)alloc((size_t)BATCH * K2 * 2),
                              (unsigned short*)alloc((size_t)BATCH * K2 * 2) };
    float* H1  = (float*)alloc((size_t)BATCH * UNITS * 4);
    float* C1  = (float*)alloc((size_t)BATCH * UNITS * 4);
    float* H2  = (float*)alloc((size_t)BATCH * UNITS * 4);
    float* C2  = (float*)alloc((size_t)BATCH * UNITS * 4);
    float* Ctx = (float*)alloc((size_t)BATCH * UNITS * 4);

    // Optional bf16 copy of the encoder memory (64MB) if workspace allows.
    const size_t hb_bytes = (size_t)BATCH * TENC * UNITS * 2;
    unsigned short* Hb = nullptr;
    bool use_hb = (ws_size >= used + hb_bytes + 256);
    if (use_hb) Hb = (unsigned short*)alloc(hb_bytes);

    // prologue: weight fuse/transpose/quantize, state init, encoder convert
    prep_w1<<<(NOUT * K1 + 255) / 256, 256, 0, stream>>>(W1, U1, W1t);
    prep_w2<<<(NOUT * K2 + 255) / 256, 256, 0, stream>>>(W2, U2, W2t);
    init_state<<<(BATCH * K1 + 255) / 256, 256, 0, stream>>>(
        X1[0], X1[1], X2[0], X2[1], C1, C2, y);
    if (use_hb)
        conv_h<<<(BATCH * TENC * UNITS) / (256 * 4), 256, 0, stream>>>(h, Hb);

    for (int t = 0; t < TDEC; ++t) {
        int cur = t & 1, nxt = cur ^ 1;
        // LSTM cell 1: reads X1[cur], writes h1 -> H1 (fp32) and X1[nxt] slot 0
        lstm_cell<<<16, 256, 0, stream>>>(X1[cur], K1, W1t, b1, C1, H1,
                                          X1[nxt], K1, 0);
        // attention: ctx -> Ctx, X1[nxt] ctx slice, X2[cur] ctx slice; y(t+1) -> X1[nxt]
        if (use_hb)
            attn_kernel<true><<<BATCH, 256, 0, stream>>>(Hb, H1, Ctx, X1[nxt], X2[cur], y, t);
        else
            attn_kernel<false><<<BATCH, 256, 0, stream>>>(h, H1, Ctx, X1[nxt], X2[cur], y, t);
        // LSTM cell 2: reads X2[cur], writes h2 -> H2 (fp32) and X2[nxt] slot 512
        lstm_cell<<<16, 256, 0, stream>>>(X2[cur], K2, W2t, b2, C2, H2,
                                          X2[nxt], K2, UNITS);
        dense_softmax<<<BATCH, 64, 0, stream>>>(H2, Ctx, Wd, bd,
                                                out + (size_t)t * BATCH * VOCAB);
    }
}
